// HierarchicalRefinementQuantizer_55886114456195
// MI455X (gfx1250) — compile-verified, matrix operand order checked by reading, not run on hardware
//
#include <hip/hip_runtime.h>

#define Bsz   4096
#define Hn    3
#define Kn    32768
#define Dd    128
#define NSPLIT 8
#define NTILES (Kn / 16)                  // 2048
#define TILES_PER_SPLIT (NTILES / NSPLIT) // 256

typedef __attribute__((ext_vector_type(16))) _Float16 v16h;
typedef __attribute__((ext_vector_type(8)))  _Float16 v8h;
typedef __attribute__((ext_vector_type(4)))  _Float16 v4h;
typedef __attribute__((ext_vector_type(8)))  float    v8f;
typedef __attribute__((ext_vector_type(4)))  int      v4i;

// Address-space-qualified pointee types for the async builtin
// (clang HIP prints AS(1) as "__device__" in diagnostics -> arg0 is AS(1) int4*)
typedef __attribute__((address_space(1))) v4i g_v4i;
typedef __attribute__((address_space(3))) v4i l_v4i;

union HFrag { v16h f; v8h h[2]; };

// CDNA5 async global->LDS staging path (ASYNCcnt), with portable fallback.
#if __has_builtin(__builtin_amdgcn_global_load_async_to_lds_b128) && \
    __has_builtin(__builtin_amdgcn_s_wait_asynccnt)
#define HRQ_ASYNC 1
#endif

__device__ __forceinline__ void hrq_stage16B(const _Float16* gsrc, _Float16* ldst) {
#ifdef HRQ_ASYNC
    __builtin_amdgcn_global_load_async_to_lds_b128((g_v4i*)gsrc, (l_v4i*)ldst, 0, 0);
#else
    *(v8h*)ldst = *(const v8h*)gsrc;
#endif
}

__device__ __forceinline__ void hrq_stage_wait() {
#ifdef HRQ_ASYNC
    __builtin_amdgcn_s_wait_asynccnt(0);
#endif
}

// ---------------------------------------------------------------------------
// Kernel 1: resid_f32 = inputs, resid_f16 = (f16)inputs, loss = 0
// ---------------------------------------------------------------------------
__global__ __launch_bounds__(256)
void hrq_init_kernel(const float* __restrict__ in, float* __restrict__ residf,
                     _Float16* __restrict__ residh, float* __restrict__ loss) {
    int i = blockIdx.x * 256 + threadIdx.x;   // grid covers Bsz*Dd exactly
    float v = in[i];
    residf[i] = v;
    residh[i] = (_Float16)v;
    if (i < Bsz) loss[i] = 0.0f;
}

// ---------------------------------------------------------------------------
// Kernel 2: emb fp32 -> f16 copy + halfnorm[r] = 0.5 * sum(W[r]^2)
// One wave per codebook row (128 floats -> 4 per lane).
// ---------------------------------------------------------------------------
__global__ __launch_bounds__(256)
void hrq_prep_emb_kernel(const float* __restrict__ emb, _Float16* __restrict__ embf16,
                         float* __restrict__ halfnorm) {
    int wid  = threadIdx.x >> 5;
    int lane = threadIdx.x & 31;
    int row  = blockIdx.x * 8 + wid;          // 0 .. Hn*Kn-1
    const float4 v = *(const float4*)(emb + (size_t)row * Dd + lane * 4);
    v4h hv;
    hv[0] = (_Float16)v.x; hv[1] = (_Float16)v.y;
    hv[2] = (_Float16)v.z; hv[3] = (_Float16)v.w;
    *(v4h*)(embf16 + (size_t)row * Dd + lane * 4) = hv;
    float s = v.x * v.x + v.y * v.y + v.z * v.z + v.w * v.w;
#pragma unroll
    for (int m = 16; m >= 1; m >>= 1) s += __shfl_xor(s, m, 32);
    if (lane == 0) halfnorm[row] = 0.5f * s;
}

// ---------------------------------------------------------------------------
// Kernel 3: fused GEMM (scores = r.W - 0.5||W||^2) + running argmax.
// Grid = 32 row-groups x NSPLIT n-splits. Block = 256 thr = 8 waves.
// Each wave owns a 16-row WMMA tile; all 8 waves share LDS-staged B tiles
// (double-buffered, async-to-LDS when available).
// ---------------------------------------------------------------------------
__global__ __launch_bounds__(256)
void hrq_dist_kernel(const _Float16* __restrict__ residh,
                     const _Float16* __restrict__ embf16,
                     const float* __restrict__ halfnorm,
                     float* __restrict__ pbest, int* __restrict__ pidx, int h) {
    __shared__ __align__(16) _Float16 sB[2][16 * Dd];          // 2 x 4KB B tiles
    __shared__ __align__(16) float sHN[TILES_PER_SPLIT * 16];  // 16KB split norms

    const int tid  = threadIdx.x;
    const int wid  = tid >> 5;
    const int lane = tid & 31;
    const int rg   = blockIdx.x >> 3;      // row group (0..31): rows rg*128 .. +127
    const int ns   = blockIdx.x & (NSPLIT - 1);
    const int row0 = rg * 128;
    const int sub  = lane & 15;
    const int kb   = (lane >> 4) * 8;      // K-offset half-pattern (A and B symmetric)
    const int t0   = ns * TILES_PER_SPLIT;

    const _Float16* embh = embf16 + (size_t)h * Kn * Dd;
    const float* hnh     = halfnorm + (size_t)h * Kn;

    // Stage this split's half-norms into LDS (4096 floats).
    {
        const float4* src = (const float4*)(hnh + t0 * 16);
        float4* dst = (float4*)sHN;
#pragma unroll
        for (int i = tid; i < (TILES_PER_SPLIT * 16) / 4; i += 256) dst[i] = src[i];
    }

    // Load A fragments: 16 rows x 128 K, 4 chunks of K=32 (v16h each)
    const _Float16* arow = residh + (size_t)(row0 + wid * 16 + sub) * Dd;
    v16h a[4];
#pragma unroll
    for (int kc = 0; kc < 4; ++kc) {
        HFrag fr;
        fr.h[0] = *(const v8h*)(arow + kc * 32 + kb);
        fr.h[1] = *(const v8h*)(arow + kc * 32 + kb + 16);
        a[kc] = fr.f;
    }

    float best[8];
    int   bidx[8];
#pragma unroll
    for (int r = 0; r < 8; ++r) { best[r] = -3.4e38f; bidx[r] = 0; }

    // Stage first tile (each thread moves one 16B chunk of the 16x128 f16 tile)
    hrq_stage16B(embh + (size_t)(t0 * 16 + (tid >> 4)) * Dd + (tid & 15) * 8,
                 &sB[0][tid * 8]);

    for (int t = 0; t < TILES_PER_SPLIT; ++t) {
        hrq_stage_wait();     // own async stores landed in LDS
        __syncthreads();      // everyone's staging visible; prior reads done
        const int cur = t & 1;
        if (t + 1 < TILES_PER_SPLIT) {  // stage next tile into other buffer
            const int n1 = (t0 + t + 1) * 16;
            hrq_stage16B(embh + (size_t)(n1 + (tid >> 4)) * Dd + (tid & 15) * 8,
                         &sB[cur ^ 1][tid * 8]);
        }
        const int col  = (t0 + t) * 16 + sub;        // this lane's code column
        const float hn = sHN[t * 16 + sub];
        v8f acc;
#pragma unroll
        for (int r = 0; r < 8; ++r) acc[r] = -hn;    // fold -0.5||w||^2 into C

        const _Float16* bp = &sB[cur][sub * Dd + kb];
#pragma unroll
        for (int kc = 0; kc < 4; ++kc) {
            HFrag fr;
            fr.h[0] = *(const v8h*)(bp + kc * 32);
            fr.h[1] = *(const v8h*)(bp + kc * 32 + 16);
            acc = __builtin_amdgcn_wmma_f32_16x16x32_f16(
                false, a[kc], false, fr.f, (short)0, acc, false, false);
        }
#pragma unroll
        for (int r = 0; r < 8; ++r) {
            if (acc[r] > best[r]) { best[r] = acc[r]; bidx[r] = col; }
        }
    }

    // Butterfly argmax across the 16 lanes sharing each row (halves independent)
#pragma unroll
    for (int m = 1; m <= 8; m <<= 1) {
#pragma unroll
        for (int r = 0; r < 8; ++r) {
            float ov = __shfl_xor(best[r], m, 32);
            int   oi = __shfl_xor(bidx[r], m, 32);
            if (ov > best[r] || (ov == best[r] && oi < bidx[r])) { best[r] = ov; bidx[r] = oi; }
        }
    }
    if (sub == 0) {  // lane 0 -> rows wid*16+0..7 ; lane 16 -> rows wid*16+8..15
        const int rbase = row0 + wid * 16 + (lane >> 4) * 8;
#pragma unroll
        for (int r = 0; r < 8; ++r) {
            pbest[(size_t)(rbase + r) * NSPLIT + ns] = best[r];
            pidx [(size_t)(rbase + r) * NSPLIT + ns] = bidx[r];
        }
    }
}

// ---------------------------------------------------------------------------
// Kernel 4: reduce the NSPLIT partial argmaxes per row -> final code
// ---------------------------------------------------------------------------
__global__ __launch_bounds__(256)
void hrq_argmax_kernel(const float* __restrict__ pbest, const int* __restrict__ pidx,
                       int* __restrict__ codes_i, float* __restrict__ d_codes, int h) {
    int row = blockIdx.x * 256 + threadIdx.x;   // grid covers Bsz exactly
    float bv = -3.4e38f;
    int   bi = 0;
#pragma unroll
    for (int ns = 0; ns < NSPLIT; ++ns) {
        float v = pbest[(size_t)row * NSPLIT + ns];
        int   i = pidx [(size_t)row * NSPLIT + ns];
        if (v > bv || (v == bv && i < bi)) { bv = v; bi = i; }
    }
    codes_i[row] = bi;
    d_codes[(size_t)row * Hn + h] = (float)bi;
}

// ---------------------------------------------------------------------------
// Kernel 5: resid -= W[code] (fp32, exact); quantized += W[code]; refresh f16
// ---------------------------------------------------------------------------
__global__ __launch_bounds__(256)
void hrq_update_kernel(const float* __restrict__ emb, const int* __restrict__ codes_i,
                       float* __restrict__ residf, _Float16* __restrict__ residh,
                       float* __restrict__ quant, int h) {
    int e = blockIdx.x * 256 + threadIdx.x;     // grid covers Bsz*Dd exactly
    int row = e >> 7;
    int col = e & 127;
    int code = codes_i[row];
    float q  = emb[(size_t)h * Kn * Dd + (size_t)code * Dd + col];
    float rv = residf[e] - q;
    residf[e] = rv;
    residh[e] = (_Float16)rv;
    float qa = (h == 0) ? q : (quant[e] + q);
    quant[e] = qa;
}

// ---------------------------------------------------------------------------
extern "C" void kernel_launch(void* const* d_in, const int* in_sizes, int n_in,
                              void* d_out, int out_size, void* d_ws, size_t ws_size,
                              hipStream_t stream) {
    const float* inputs = (const float*)d_in[0];   // (B,1,D) fp32
    const float* emb    = (const float*)d_in[1];   // (H,K,D) fp32

    float* loss    = (float*)d_out;                // [Bsz]
    float* quant   = loss + Bsz;                   // [Bsz*Dd]
    float* d_codes = quant + (size_t)Bsz * Dd;     // [Bsz*Hn] (codes as floats)

    char* ws = (char*)d_ws;
    size_t off = 0;
    _Float16* embf16   = (_Float16*)(ws + off); off += (size_t)Hn * Kn * Dd * sizeof(_Float16);
    float*    halfnorm = (float*)(ws + off);    off += (size_t)Hn * Kn * sizeof(float);
    float*    residf   = (float*)(ws + off);    off += (size_t)Bsz * Dd * sizeof(float);
    _Float16* residh   = (_Float16*)(ws + off); off += (size_t)Bsz * Dd * sizeof(_Float16);
    float*    pbest    = (float*)(ws + off);    off += (size_t)Bsz * NSPLIT * sizeof(float);
    int*      pidx     = (int*)(ws + off);      off += (size_t)Bsz * NSPLIT * sizeof(int);
    int*      codes_i  = (int*)(ws + off);      off += (size_t)Bsz * sizeof(int);

    hrq_init_kernel<<<(Bsz * Dd) / 256, 256, 0, stream>>>(inputs, residf, residh, loss);
    hrq_prep_emb_kernel<<<(Hn * Kn) / 8, 256, 0, stream>>>(emb, embf16, halfnorm);

    for (int h = 0; h < Hn; ++h) {
        hrq_dist_kernel<<<32 * NSPLIT, 256, 0, stream>>>(residh, embf16, halfnorm, pbest, pidx, h);
        hrq_argmax_kernel<<<Bsz / 256, 256, 0, stream>>>(pbest, pidx, codes_i, d_codes, h);
        hrq_update_kernel<<<(Bsz * Dd) / 256, 256, 0, stream>>>(emb, codes_i, residf, residh, quant, h);
    }
}
